// Head_70033736728678
// MI455X (gfx1250) — matrix operand order repaired and checked by reading, not compiled, not run on hardware
//
#include <hip/hip_runtime.h>
#include <hip/hip_bf16.h>

typedef __attribute__((ext_vector_type(16))) _Float16 v16h;
typedef __attribute__((ext_vector_type(8)))  float    v8f;
typedef __attribute__((ext_vector_type(4)))  unsigned int v4u;
typedef __attribute__((ext_vector_type(4)))  float    v4f;

#define TSEQ 256
#define HS   64
#define XP   72    // pitch (halves) for Qm / Km rows
#define VP   264   // pitch (halves) for Vt rows
#define PP   40    // pitch (halves) for per-wave P rows
#define NWAVE 16
#define NTHREADS 512

struct SMem {
    _Float16 Qm[TSEQ * XP];        // Q row-major f16
    _Float16 Km[TSEQ * XP];        // K row-major f16
    _Float16 Vt[HS * VP];          // V transposed: [h][t]
    _Float16 Wt[3 * HS * HS];      // W transposed: [proj][out][in]
    _Float16 P[NWAVE * 16 * PP];   // per-wave P tile (16 x 32 used)
    float    bias[3 * HS];
};
// sizeof(SMem) ~= 153.3 KB -> 2 workgroups resident per 320KB WGP

union FragU { v16h h; _Float16 e[16]; v4u u[2]; };

__device__ __forceinline__ v8f wmma16(v16h a, v16h b, v8f c) {
    return __builtin_amdgcn_wmma_f32_16x16x32_f16(
        false, a, false, b, (short)0, c, false, false);
}

// A-fragment (16x32 f16) from row-major f16 LDS array with pitch.
// elem e<8  -> K = e + 8g + koff ; elem e>=8 -> K = (e-8) + 16 + 8g + koff
__device__ __forceinline__ v16h loadA(const _Float16* M, int rowbase, int pitch,
                                      int koff, int ln, int g) {
    FragU f;
    const _Float16* p = M + (rowbase + ln) * pitch + koff + 8 * g;
    f.u[0] = *(const v4u*)p;
    f.u[1] = *(const v4u*)(p + 16);
    return f.h;
}

// A-fragment directly from global fp32 (non-temporal), converted to f16.
// p points at X[row] + koff + 8g (floats). Elements per layout above.
__device__ __forceinline__ v16h loadA_xf32(const float* p) {
    FragU f;
    v4f a = __builtin_nontemporal_load((const v4f*)(p));
    v4f b = __builtin_nontemporal_load((const v4f*)(p + 4));
    v4f c = __builtin_nontemporal_load((const v4f*)(p + 16));
    v4f d = __builtin_nontemporal_load((const v4f*)(p + 20));
    #pragma unroll
    for (int k = 0; k < 4; ++k) {
        f.e[k]      = (_Float16)a[k];
        f.e[4 + k]  = (_Float16)b[k];
        f.e[8 + k]  = (_Float16)c[k];
        f.e[12 + k] = (_Float16)d[k];
    }
    return f.h;
}

// B-fragment (32x16 f16): B(k,n) = M[(nbase+n)*pitch + koff + k]
// lane ln = column n; elem e<8 -> K = e + 8g ; elem e>=8 -> K = (e-8) + 16 + 8g.
__device__ __forceinline__ v16h loadB(const _Float16* M, int nbase, int pitch,
                                      int koff, int ln, int g) {
    FragU f;
    const _Float16* p = M + (nbase + ln) * pitch + koff + 8 * g;
    f.u[0] = *(const v4u*)p;
    f.u[1] = *(const v4u*)(p + 16);
    return f.h;
}

extern __shared__ __align__(16) char smem_raw[];

__global__ void __launch_bounds__(NTHREADS)
attn_head_kernel(const float* __restrict__ X,
                 const float* __restrict__ Wk, const float* __restrict__ bk,
                 const float* __restrict__ Wq, const float* __restrict__ bq,
                 const float* __restrict__ Wv, const float* __restrict__ bv,
                 float* __restrict__ Out)
{
    SMem& S = *(SMem*)smem_raw;
    const int tid  = threadIdx.x;
    const int wid  = tid >> 5;
    const int lane = tid & 31;
    const int g    = lane >> 4;     // lane group (0/1)
    const int ln   = lane & 15;     // lane within group
    const int b    = blockIdx.x;

    // ---------------- Phase 0: stage transposed weights + bias --------------
    {
        const float* Wg[3] = {Wk, Wq, Wv};
        const float* Bg[3] = {bk, bq, bv};
        #pragma unroll
        for (int p = 0; p < 3; ++p) {
            #pragma unroll
            for (int it = 0; it < 8; ++it) {
                int idx = it * NTHREADS + tid;   // idx = e*64 + o
                float w = Wg[p][idx];
                int e = idx >> 6, o = idx & 63;
                S.Wt[p * HS * HS + o * HS + e] = (_Float16)w;
            }
            if (tid < HS) S.bias[p * HS + tid] = Bg[p][tid];
        }
    }

    // ---------------- Phase 1: K/Q/V projections via WMMA -------------------
    const int r0 = wid * 16;   // this wave's 16 rows
    // X fragments straight from global (each element read exactly once)
    const float* Xrow = X + (size_t)b * TSEQ * HS + (size_t)(r0 + ln) * HS;
    v16h ax0 = loadA_xf32(Xrow + 8 * g);
    v16h ax1 = loadA_xf32(Xrow + 32 + 8 * g);

    __syncthreads();   // weights staged

    #pragma unroll
    for (int p = 0; p < 3; ++p) {
        #pragma unroll
        for (int nt = 0; nt < 4; ++nt) {
            v16h b0 = loadB(&S.Wt[p * HS * HS], 16 * nt, HS, 0,  ln, g);
            v16h b1 = loadB(&S.Wt[p * HS * HS], 16 * nt, HS, 32, ln, g);
            v8f c = {};
            c = wmma16(ax0, b0, c);
            c = wmma16(ax1, b1, c);
            float bb = S.bias[p * HS + 16 * nt + ln];
            if (p == 0) {
                #pragma unroll
                for (int v = 0; v < 8; ++v)
                    S.Km[(r0 + v + 8 * g) * XP + 16 * nt + ln] = (_Float16)(c[v] + bb);
            } else if (p == 1) {
                #pragma unroll
                for (int v = 0; v < 8; ++v)
                    S.Qm[(r0 + v + 8 * g) * XP + 16 * nt + ln] = (_Float16)(c[v] + bb);
            } else {
                #pragma unroll
                for (int v = 0; v < 8; ++v)
                    S.Vt[(16 * nt + ln) * VP + r0 + v + 8 * g] = (_Float16)(c[v] + bb);
            }
        }
    }
    __syncthreads();

    // ---------------- Phase 2: flash attention, wave i = query tile --------
    const int i = wid;
    v16h aq0 = loadA(S.Qm, r0, XP, 0,  ln, g);
    v16h aq1 = loadA(S.Qm, r0, XP, 32, ln, g);

    v8f o0 = {}, o1 = {}, o2 = {}, o3 = {};
    float m[8], l[8];
    #pragma unroll
    for (int v = 0; v < 8; ++v) { m[v] = -1e30f; l[v] = 0.0f; }

    _Float16* Pw = &S.P[wid * 16 * PP];
    const float scale = 0.125f;     // 1/sqrt(64)
    const int npairs = (i + 2) >> 1;

    for (int jp = 0; jp < npairs; ++jp) {
        float s[2][8];
        #pragma unroll
        for (int js = 0; js < 2; ++js) {
            int j = 2 * jp + js;
            if (j <= i) {                       // wave-uniform branch
                v16h bk0 = loadB(S.Km, 16 * j, XP, 0,  ln, g);
                v16h bk1 = loadB(S.Km, 16 * j, XP, 32, ln, g);
                v8f c = {};
                c = wmma16(aq0, bk0, c);
                c = wmma16(aq1, bk1, c);
                #pragma unroll
                for (int v = 0; v < 8; ++v) {
                    float val = c[v] * scale;
                    if (j == i && ln > v + 8 * g) val = -1e30f;  // causal
                    s[js][v] = val;
                }
            } else {
                #pragma unroll
                for (int v = 0; v < 8; ++v) s[js][v] = -1e30f;
            }
        }
        // online softmax: row stats across the 16 lanes holding each row
        #pragma unroll
        for (int v = 0; v < 8; ++v) {
            float r = fmaxf(s[0][v], s[1][v]);
            #pragma unroll
            for (int d = 1; d < 16; d <<= 1) r = fmaxf(r, __shfl_xor(r, d, 16));
            float mn = fmaxf(m[v], r);
            float p0 = __expf(s[0][v] - mn);
            float p1 = __expf(s[1][v] - mn);
            Pw[(v + 8 * g) * PP + ln]      = (_Float16)p0;
            Pw[(v + 8 * g) * PP + 16 + ln] = (_Float16)p1;
            float rs = p0 + p1;
            #pragma unroll
            for (int d = 1; d < 16; d <<= 1) rs += __shfl_xor(rs, d, 16);
            float co = __expf(m[v] - mn);
            l[v] = l[v] * co + rs;
            m[v] = mn;
            o0[v] *= co; o1[v] *= co; o2[v] *= co; o3[v] *= co;
        }
        // P (16x32) as A-fragment, V tiles as B-fragments
        v16h ap = loadA(Pw, 0, PP, 0, ln, g);
        v16h bv0 = loadB(S.Vt, 0,  VP, 32 * jp, ln, g);
        o0 = wmma16(ap, bv0, o0);
        v16h bv1 = loadB(S.Vt, 16, VP, 32 * jp, ln, g);
        o1 = wmma16(ap, bv1, o1);
        v16h bv2 = loadB(S.Vt, 32, VP, 32 * jp, ln, g);
        o2 = wmma16(ap, bv2, o2);
        v16h bv3 = loadB(S.Vt, 48, VP, 32 * jp, ln, g);
        o3 = wmma16(ap, bv3, o3);
    }

    // ---------------- Epilogue: normalize + store fp32 (non-temporal) -------
    float* Og = Out + (size_t)b * TSEQ * HS;
    #pragma unroll
    for (int v = 0; v < 8; ++v) {
        float inv = 1.0f / l[v];
        int row = r0 + v + 8 * g;
        __builtin_nontemporal_store(o0[v] * inv, &Og[row * HS + 0  + ln]);
        __builtin_nontemporal_store(o1[v] * inv, &Og[row * HS + 16 + ln]);
        __builtin_nontemporal_store(o2[v] * inv, &Og[row * HS + 32 + ln]);
        __builtin_nontemporal_store(o3[v] * inv, &Og[row * HS + 48 + ln]);
    }
}

extern "C" void kernel_launch(void* const* d_in, const int* in_sizes, int n_in,
                              void* d_out, int out_size, void* d_ws, size_t ws_size,
                              hipStream_t stream) {
    const float* X  = (const float*)d_in[0];
    const float* Wk = (const float*)d_in[1];
    const float* bk = (const float*)d_in[2];
    const float* Wq = (const float*)d_in[3];
    const float* bq = (const float*)d_in[4];
    const float* Wv = (const float*)d_in[5];
    const float* bv = (const float*)d_in[6];
    float* Out = (float*)d_out;

    const size_t smem = sizeof(SMem);
    (void)hipFuncSetAttribute((const void*)attn_head_kernel,
                              hipFuncAttributeMaxDynamicSharedMemorySize,
                              (int)smem);
    const int B = 4096;
    attn_head_kernel<<<B, NTHREADS, smem, stream>>>(X, Wk, bk, Wq, bq, Wv, bv, Out);
}
